// GraphSAGE_37443524886927
// MI455X (gfx1250) — matrix operand order repaired and checked by reading, not compiled, not run on hardware
//
#include <hip/hip_runtime.h>
#include <hip/hip_bf16.h>

// ---------------------------------------------------------------------------
// GraphSAGE on MI455X (gfx1250, wave32, WMMA)
//   agg phase: f32 global atomics (L2-resident: x=51MB, agg=51MB << 192MB L2)
//   GEMM phase: v_wmma_f32_16x16x32_f16, f32 accumulate, f16 activations
//   Epilogues: block-uniform fast path -> unmasked stores for 781/782 blocks
// ---------------------------------------------------------------------------

#define N_NODES 100000
#define N_EDGES 1600000
#define F_IN    128
#define HID     128
#define N_CLS   40
#define N_CLS_P 48   // padded to 3 WMMA column tiles

typedef __attribute__((ext_vector_type(16))) _Float16 v16h;
typedef __attribute__((ext_vector_type(8)))  _Float16 v8h;
typedef __attribute__((ext_vector_type(8)))  float    v8f;
typedef __attribute__((ext_vector_type(4)))  float    v4f;

// ---- small load helpers ----------------------------------------------------
__device__ __forceinline__ v8f load8(const float* p) {
    const v4f* q = (const v4f*)p;
    v4f a = q[0], b = q[1];
    v8f r; r[0]=a[0]; r[1]=a[1]; r[2]=a[2]; r[3]=a[3];
           r[4]=b[0]; r[5]=b[1]; r[6]=b[2]; r[7]=b[3];
    return r;
}
__device__ __forceinline__ v8f load8(const _Float16* p) {
    v8h h = *(const v8h*)p;
    v8f r;
#pragma unroll
    for (int j = 0; j < 8; ++j) r[j] = (float)h[j];
    return r;
}

// ---- weight transpose + f32->f16 convert -----------------------------------
__global__ void prep_weights(const float* __restrict__ W1l, const float* __restrict__ W1r,
                             const float* __restrict__ W2l, const float* __restrict__ W2r,
                             const float* __restrict__ Wout,
                             _Float16* __restrict__ T1l, _Float16* __restrict__ T1r,
                             _Float16* __restrict__ T2l, _Float16* __restrict__ T2r,
                             _Float16* __restrict__ Tout)
{
    int idx = blockIdx.x * blockDim.x + threadIdx.x;
    if (idx < 128 * 128) {
        int k = idx >> 7, n = idx & 127;          // W[k][n] row-major
        int t = n * 128 + k;                      // T[n][k]
        T1l[t] = (_Float16)W1l[idx];
        T1r[t] = (_Float16)W1r[idx];
        T2l[t] = (_Float16)W2l[idx];
        T2r[t] = (_Float16)W2r[idx];
    }
    if (idx < N_CLS_P * 128) {
        int n = idx >> 7, k = idx & 127;
        Tout[n * 128 + k] = (n < N_CLS) ? (_Float16)Wout[k * N_CLS + n] : (_Float16)0.0f;
    }
}

// ---- degree counts ---------------------------------------------------------
__global__ void degree_kernel(const long long* __restrict__ dst, float* __restrict__ cnt, int nE)
{
    int e = blockIdx.x * blockDim.x + threadIdx.x;
    if (e < nE) atomicAdd(&cnt[(int)dst[e]], 1.0f);
}

// ---- edge gather + atomic scatter: one wave per edge, 32 lanes x float4 ----
template <typename T>
__global__ void scatter_kernel(const T* __restrict__ feat, const long long* __restrict__ src,
                               const long long* __restrict__ dst, float* __restrict__ agg, int nE)
{
    int e    = (int)((blockIdx.x * blockDim.x + threadIdx.x) >> 5);
    int lane = threadIdx.x & 31;
    if (e >= nE) return;
    int s = __builtin_amdgcn_readfirstlane((int)src[e]);
    int d = __builtin_amdgcn_readfirstlane((int)dst[e]);
    const T* p = feat + (size_t)s * 128 + lane * 4;
    float*   q = agg  + (size_t)d * 128 + lane * 4;
    float v0 = (float)p[0], v1 = (float)p[1], v2 = (float)p[2], v3 = (float)p[3];
    atomicAdd(q + 0, v0);
    atomicAdd(q + 1, v1);
    atomicAdd(q + 2, v2);
    atomicAdd(q + 3, v3);
}

// ---- SAGE layer GEMM: out = act( (agg/deg) @ Wl + x @ Wr + b ) -------------
// 256 threads = 8 waves; wave -> 16 rows; block -> 128 rows.
// A layout (16x32 f16): lane = row (lane&15), half-wave selects K group of 16.
// B layout (32x16 f16): lane = col (lane&15), half-wave selects K group of 16.
template <typename TIN, typename TOUT, bool RELU>
__global__ void sage_gemm(const float* __restrict__ agg, const float* __restrict__ cnt,
                          const TIN* __restrict__ xin,
                          const _Float16* __restrict__ WlT, const _Float16* __restrict__ WrT,
                          const float* __restrict__ bias, TOUT* __restrict__ out, int nRows)
{
    __shared__ __align__(32) _Float16 sWl[128 * 128];
    __shared__ __align__(32) _Float16 sWr[128 * 128];
    {
        const uint4* gl = (const uint4*)WlT;
        const uint4* gr = (const uint4*)WrT;
        uint4* ll = (uint4*)sWl;
        uint4* lr = (uint4*)sWr;
        for (int i = threadIdx.x; i < 128 * 128 / 8; i += blockDim.x) { ll[i] = gl[i]; lr[i] = gr[i]; }
    }
    __syncthreads();

    const int lane    = threadIdx.x & 31;
    const int wave    = threadIdx.x >> 5;
    const int rowBase = blockIdx.x * 128 + wave * 16;
    const int mA      = lane & 15;
    const int hiA     = lane >> 4;
    const bool fullBlock = (rowBase + 16) <= nRows;       // wave-uniform
    int rg = rowBase + mA; if (rg >= nRows) rg = nRows - 1;   // clamp loads

    const float inv = 1.0f / fmaxf(cnt[rg], 1.0f);

    // A fragments for 4 K-chunks of 32
    v16h aL[4], aR[4];
    const float* pa = agg + (size_t)rg * 128;
    const TIN*   px = xin + (size_t)rg * 128;
#pragma unroll
    for (int c = 0; c < 4; ++c) {
        const int kb = 32 * c + hiA * 16;
        v8f l0 = load8(pa + kb), l1 = load8(pa + kb + 8);
        v8f r0 = load8(px + kb), r1 = load8(px + kb + 8);
#pragma unroll
        for (int j = 0; j < 8; ++j) {
            aL[c][j]     = (_Float16)(l0[j] * inv);
            aL[c][8 + j] = (_Float16)(l1[j] * inv);
            aR[c][j]     = (_Float16)r0[j];
            aR[c][8 + j] = (_Float16)r1[j];
        }
    }

    const int nB  = lane & 15;
    const int kbB = hiA * 16;
#pragma unroll
    for (int t = 0; t < 8; ++t) {
        const int col = t * 16 + nB;
        const float bv = bias[col];
        v8f acc;
#pragma unroll
        for (int r = 0; r < 8; ++r) acc[r] = bv;
#pragma unroll
        for (int c = 0; c < 4; ++c) {
            v16h bL = *(const v16h*)&sWl[col * 128 + 32 * c + kbB];
            v16h bR = *(const v16h*)&sWr[col * 128 + 32 * c + kbB];
            acc = __builtin_amdgcn_wmma_f32_16x16x32_f16(false, aL[c], false, bL, (short)0, acc, false, false);
            acc = __builtin_amdgcn_wmma_f32_16x16x32_f16(false, aR[c], false, bR, (short)0, acc, false, false);
        }
        TOUT* po = out + (size_t)(rowBase + hiA * 8) * 128 + col;
        if (fullBlock) {                                   // scalar branch, no exec churn
#pragma unroll
            for (int r = 0; r < 8; ++r) {
                float v = acc[r];
                if (RELU) v = fmaxf(v, 0.0f);
                po[(size_t)r * 128] = (TOUT)v;
            }
        } else {
#pragma unroll
            for (int r = 0; r < 8; ++r) {
                if (rowBase + r + hiA * 8 < nRows) {
                    float v = acc[r];
                    if (RELU) v = fmaxf(v, 0.0f);
                    po[(size_t)r * 128] = (TOUT)v;
                }
            }
        }
    }
}

// ---- final 128 -> 40 GEMM (padded to 48 cols) ------------------------------
__global__ void out_gemm(const _Float16* __restrict__ h, const _Float16* __restrict__ WoT,
                         const float* __restrict__ bout, float* __restrict__ out, int nRows)
{
    __shared__ __align__(32) _Float16 sW[N_CLS_P * 128];
    for (int i = threadIdx.x; i < N_CLS_P * 128 / 8; i += blockDim.x)
        ((uint4*)sW)[i] = ((const uint4*)WoT)[i];
    __syncthreads();

    const int lane    = threadIdx.x & 31;
    const int wave    = threadIdx.x >> 5;
    const int rowBase = blockIdx.x * 128 + wave * 16;
    const int mA      = lane & 15;
    const int hiA     = lane >> 4;
    const bool fullBlock = (rowBase + 16) <= nRows;       // wave-uniform
    int rg = rowBase + mA; if (rg >= nRows) rg = nRows - 1;

    v16h aH[4];
    const _Float16* ph = h + (size_t)rg * 128;
#pragma unroll
    for (int c = 0; c < 4; ++c) {
        const int kb = 32 * c + hiA * 16;
        v8h h0 = *(const v8h*)(ph + kb);
        v8h h1 = *(const v8h*)(ph + kb + 8);
#pragma unroll
        for (int j = 0; j < 8; ++j) { aH[c][j] = h0[j]; aH[c][8 + j] = h1[j]; }
    }

    const int nB  = lane & 15;
    const int kbB = hiA * 16;
#pragma unroll
    for (int t = 0; t < 3; ++t) {
        const int col = t * 16 + nB;
        const bool colOK = (t < 2) || (col < N_CLS);      // only tile 2 needs col mask
        const float bv = colOK ? bout[col] : 0.0f;
        v8f acc;
#pragma unroll
        for (int r = 0; r < 8; ++r) acc[r] = bv;
#pragma unroll
        for (int c = 0; c < 4; ++c) {
            v16h b = *(const v16h*)&sW[col * 128 + 32 * c + kbB];
            acc = __builtin_amdgcn_wmma_f32_16x16x32_f16(false, aH[c], false, b, (short)0, acc, false, false);
        }
        float* po = out + (size_t)(rowBase + hiA * 8) * N_CLS + col;
        if (t < 2 && fullBlock) {                          // branch-free store path
#pragma unroll
            for (int r = 0; r < 8; ++r) po[(size_t)r * N_CLS] = acc[r];
        } else {
#pragma unroll
            for (int r = 0; r < 8; ++r) {
                if ((rowBase + r + hiA * 8 < nRows) && colOK)
                    po[(size_t)r * N_CLS] = acc[r];
            }
        }
    }
}

// ---------------------------------------------------------------------------
extern "C" void kernel_launch(void* const* d_in, const int* in_sizes, int n_in,
                              void* d_out, int out_size, void* d_ws, size_t ws_size,
                              hipStream_t stream)
{
    const float*     x    = (const float*)d_in[0];
    const long long* ei   = (const long long*)d_in[1];   // int64 [2, E]
    const float*     W1l  = (const float*)d_in[2];
    const float*     W1r  = (const float*)d_in[3];
    const float*     b1   = (const float*)d_in[4];
    const float*     W2l  = (const float*)d_in[5];
    const float*     W2r  = (const float*)d_in[6];
    const float*     b2   = (const float*)d_in[7];
    const float*     Wout = (const float*)d_in[8];
    const float*     bout = (const float*)d_in[9];
    const long long* src  = ei;
    const long long* dst  = ei + N_EDGES;

    // workspace layout (bytes)
    char* ws = (char*)d_ws;
    const size_t AGG_B = (size_t)N_NODES * 128 * 4;      // 51.2 MB f32
    const size_t CNT_B = (size_t)N_NODES * 4;            // 0.4 MB f32
    const size_t H_B   = (size_t)N_NODES * 128 * 2;      // 25.6 MB f16
    const size_t W_B   = 128 * 128 * 2;                  // 32 KB f16
    float*     agg = (float*)(ws);
    float*     cnt = (float*)(ws + AGG_B);
    _Float16*  h1  = (_Float16*)(ws + AGG_B + CNT_B);
    _Float16*  h2  = (_Float16*)(ws + AGG_B + CNT_B + H_B);
    _Float16*  T1l = (_Float16*)(ws + AGG_B + CNT_B + 2 * H_B);
    _Float16*  T1r = (_Float16*)(ws + AGG_B + CNT_B + 2 * H_B + 1 * W_B);
    _Float16*  T2l = (_Float16*)(ws + AGG_B + CNT_B + 2 * H_B + 2 * W_B);
    _Float16*  T2r = (_Float16*)(ws + AGG_B + CNT_B + 2 * H_B + 3 * W_B);
    _Float16*  Tout= (_Float16*)(ws + AGG_B + CNT_B + 2 * H_B + 4 * W_B);

    const int gemmGrid    = (N_NODES + 127) / 128;       // 782
    const int scatterGrid = (N_EDGES + 7) / 8;           // 8 edges per 256-thr block

    (void)hipMemsetAsync(agg, 0, AGG_B, stream);
    (void)hipMemsetAsync(cnt, 0, CNT_B, stream);

    prep_weights<<<(128 * 128 + 255) / 256, 256, 0, stream>>>(W1l, W1r, W2l, W2r, Wout,
                                                              T1l, T1r, T2l, T2r, Tout);
    degree_kernel<<<(N_EDGES + 255) / 256, 256, 0, stream>>>(dst, cnt, N_EDGES);

    // layer 1
    scatter_kernel<float><<<scatterGrid, 256, 0, stream>>>(x, src, dst, agg, N_EDGES);
    sage_gemm<float, _Float16, true><<<gemmGrid, 256, 0, stream>>>(agg, cnt, x, T1l, T1r, b1, h1, N_NODES);

    // layer 2
    (void)hipMemsetAsync(agg, 0, AGG_B, stream);
    scatter_kernel<_Float16><<<scatterGrid, 256, 0, stream>>>(h1, src, dst, agg, N_EDGES);
    sage_gemm<_Float16, _Float16, true><<<gemmGrid, 256, 0, stream>>>(agg, cnt, h1, T2l, T2r, b2, h2, N_NODES);

    // classifier
    out_gemm<<<gemmGrid, 256, 0, stream>>>(h2, Tout, bout, (float*)d_out, N_NODES);
}
